// LlamaAttention_28028956573830
// MI455X (gfx1250) — matrix operand order repaired and checked by reading, not compiled, not run on hardware
//
#include <hip/hip_runtime.h>
#include <hip/hip_bf16.h>

// ---------------- types ----------------
typedef __bf16 v16bf __attribute__((ext_vector_type(16)));
typedef float  v8f   __attribute__((ext_vector_type(8)));
typedef int    v4i   __attribute__((ext_vector_type(4)));

union FragAB { v16bf v; uint4 q[2]; unsigned short h[16]; };
union FragC  { v8f v; float f[8]; };

#define NB 4
#define SS 1024
#define HH 2048
#define NHD 16
#define HD 128
#define TT 8
#define RR 16

// LDS tile row strides (elements), padded for bank spread, 16B-aligned rows
#define BROW 40    // GEMM B-tile rows: 32 K elems + 8 pad
#define KROW 136   // flash K-tile rows: 128 d elems + 8 pad
#define VROW 40    // flash V^T-tile rows: 32 key elems + 8 pad

#if defined(__has_builtin)
#if __has_builtin(__builtin_amdgcn_global_load_async_to_lds_b128) && \
    __has_builtin(__builtin_amdgcn_s_wait_asynccnt)
#define USE_ASYNC_LDS 1
#endif
#endif
#ifndef USE_ASYNC_LDS
#define USE_ASYNC_LDS 0
#endif

#if USE_ASYNC_LDS
#define ASYNC_COPY16(g, l) \
    __builtin_amdgcn_global_load_async_to_lds_b128((v4i*)(g), (v4i*)(l), 0, 0)
#define ASYNC_WAIT() __builtin_amdgcn_s_wait_asynccnt(0)
#else
#define ASYNC_COPY16(g, l) (*(uint4*)(l) = *(const uint4*)(g))
#define ASYNC_WAIT()
#endif

__device__ __forceinline__ unsigned short f2bf(float f) {
    unsigned u = __float_as_uint(f);
    unsigned r = u + 0x7FFFu + ((u >> 16) & 1u);
    return (unsigned short)(r >> 16);
}
__device__ __forceinline__ float bf2f(unsigned short h) {
    return __uint_as_float(((unsigned)h) << 16);
}

// A-fragment (16x32 bf16): lane m=lane&15, hi=lane>>4; elem e -> K = hi*8+e (+16 for e>=8)
__device__ __forceinline__ void load_a(FragAB& f, const unsigned short* base, int ld,
                                       int row0, int k0, int lane) {
    int m = lane & 15, hi = lane >> 4;
    const uint4* p = (const uint4*)(base + (size_t)(row0 + m) * ld + k0 + hi * 8);
    f.q[0] = p[0];
    f.q[1] = p[2];
}
// B-fragment (32x16 bf16) from row-major (N x K): lane n=lane&15, hi=lane>>4; elem e -> K = hi*8+e (+16 for e>=8)
__device__ __forceinline__ void load_b(FragAB& f, const unsigned short* base, int ld,
                                       int n0, int k0, int lane) {
    int n = lane & 15, hi = lane >> 4;
    const uint4* p = (const uint4*)(base + (size_t)(n0 + n) * ld + k0 + hi * 8);
    f.q[0] = p[0];
    f.q[1] = p[2];
}
// B-fragment from an LDS tile with row stride ld (elements)
__device__ __forceinline__ void load_b_lds(FragAB& f, const unsigned short* base, int ld,
                                           int n0, int k0, int lane) {
    int n = lane & 15, hi = lane >> 4;
    const uint4* p = (const uint4*)(base + (n0 + n) * ld + k0 + hi * 8);
    f.q[0] = p[0];
    f.q[1] = p[2];
}

#define WMMA_BF16(A, B, C) \
    __builtin_amdgcn_wmma_f32_16x16x32_bf16(false, (A).v, false, (B).v, (short)0, (C).v, false, false)

// cooperative stage of one 128(N)x32(K) B slab into LDS (4 x b128 per thread, 128 threads)
__device__ __forceinline__ void stage_btile(unsigned short* buf, const unsigned short* W,
                                            int n0, int k0, int tid) {
#pragma unroll
    for (int c = 0; c < 4; ++c) {
        int chunk = tid + 128 * c;          // 0..511
        int row = chunk >> 2;               // 0..127
        int k8 = (chunk & 3) * 8;           // 0,8,16,24
        ASYNC_COPY16(W + (size_t)(n0 + row) * HH + k0 + k8, buf + row * BROW + k8);
    }
}
// stage 32 key rows x 128 d of K into LDS
__device__ __forceinline__ void stage_ktile(unsigned short* buf, const unsigned short* kbase,
                                            int j0, int tid) {
#pragma unroll
    for (int c = 0; c < 4; ++c) {
        int chunk = tid + 128 * c;          // 0..511
        int row = chunk >> 4;               // 0..31 (key)
        int k8 = (chunk & 15) * 8;          // 0..120 (d)
        ASYNC_COPY16(kbase + (size_t)(j0 + row) * HD + k8, buf + row * KROW + k8);
    }
}
// stage 128 d rows x 32 keys of V^T into LDS
__device__ __forceinline__ void stage_vtile(unsigned short* buf, const unsigned short* vbase,
                                            int j0, int tid) {
#pragma unroll
    for (int c = 0; c < 4; ++c) {
        int chunk = tid + 128 * c;          // 0..511
        int row = chunk >> 2;               // 0..127 (d)
        int k8 = (chunk & 3) * 8;           // 0..24 (key)
        ASYNC_COPY16(vbase + (size_t)row * SS + j0 + k8, buf + row * VROW + k8);
    }
}

// ---------------- kernel 1: gating (feat mean + gaussian logits + softmax top2) ----------------
__global__ void k_gate(const float* __restrict__ x,
                       const float* __restrict__ mq, const float* __restrict__ vq,
                       const float* __restrict__ mv, const float* __restrict__ vv,
                       float* __restrict__ wq, float* __restrict__ wvout) {
    __shared__ float sfeat[HH];
    __shared__ float slog[16];
    int b = blockIdx.x;
    for (int h = threadIdx.x; h < HH; h += blockDim.x) {
        float s = 0.f;
        const float* px = x + ((size_t)b * SS) * HH + h;
        for (int t = 0; t < SS; ++t) s += px[(size_t)t * HH];
        sfeat[h] = s * (1.0f / (float)SS);
    }
    if (threadIdx.x < 16) slog[threadIdx.x] = 0.f;
    __syncthreads();
    float acc[16];
    for (int t = 0; t < 16; ++t) acc[t] = 0.f;
    const float LOG2PI = 1.8378770664093453f;
    for (int h = threadIdx.x; h < HH; h += blockDim.x) {
        float f = sfeat[h];
#pragma unroll
        for (int t = 0; t < TT; ++t) {
            float va = vq[t * HH + h] + 1e-6f;
            float d = f - mq[t * HH + h];
            acc[t] += -0.5f * (LOG2PI + __logf(va) + d * d / va);
            float vb = vv[t * HH + h] + 1e-6f;
            float d2 = f - mv[t * HH + h];
            acc[8 + t] += -0.5f * (LOG2PI + __logf(vb) + d2 * d2 / vb);
        }
    }
    for (int t = 0; t < 16; ++t) atomicAdd(&slog[t], acc[t]);
    __syncthreads();
    if (threadIdx.x == 0) {
        for (int g = 0; g < 2; ++g) {
            float* out = g ? wvout : wq;
            float l[TT];
            for (int t = 0; t < TT; ++t) l[t] = slog[g * 8 + t];
            float mx = l[0];
            for (int t = 1; t < TT; ++t) mx = fmaxf(mx, l[t]);
            float e[TT], s = 0.f;
            for (int t = 0; t < TT; ++t) { e[t] = __expf(l[t] - mx); s += e[t]; }
            float w8[TT];
            for (int t = 0; t < TT; ++t) w8[t] = e[t] / s;
            float t1 = -1.f, t2 = -1.f;
            for (int t = 0; t < TT; ++t) {
                float w = w8[t];
                if (w > t1) { t2 = t1; t1 = w; } else if (w > t2) t2 = w;
            }
            float s2 = 0.f;
            for (int t = 0; t < TT; ++t) if (w8[t] >= t2) s2 += w8[t];
            for (int t = 0; t < TT; ++t) out[b * TT + t] = (w8[t] >= t2) ? w8[t] / s2 : 0.f;
        }
    }
}

// ---------------- kernel 2: fp32 -> bf16 pack ----------------
__global__ void k_pack(const float* __restrict__ src, unsigned short* __restrict__ dst, int n) {
    int i = blockIdx.x * blockDim.x + threadIdx.x;
    int stride = gridDim.x * blockDim.x;
    for (; i < n; i += stride) dst[i] = f2bf(src[i]);
}

// ---------------- kernel 3: build concatenated LoRA matrices ----------------
__global__ void k_build_cats(const float* lAq, const float* pAq, const float* lBq, const float* pBq,
                             const float* lAv, const float* pAv, const float* lBv, const float* pBv,
                             unsigned short* Aq, unsigned short* Av,
                             unsigned short* Bq, unsigned short* Bv) {
    int which = blockIdx.y;
    int idx = blockIdx.x * blockDim.x + threadIdx.x;  // 128*2048 elems
    if (which < 2) {
        const float* lA = which ? lAv : lAq;
        const float* pA = which ? pAv : pAq;
        unsigned short* dst = which ? Av : Aq;
        int j = idx >> 11, h = idx & (HH - 1);
        int t = j >> 4, r = j & 15;
        float v = (t == 0) ? lA[r * HH + h] : pA[((size_t)(t - 1) * RR + r) * HH + h];
        dst[(size_t)j * HH + h] = f2bf(v);
    } else {
        const float* lB = (which == 2) ? lBq : lBv;
        const float* pB = (which == 2) ? pBq : pBv;
        unsigned short* dst = (which == 2) ? Bq : Bv;
        int d = idx >> 7, j = idx & 127;
        int t = j >> 4, r = j & 15;
        float v = (t == 0) ? lB[d * RR + r] : pB[((size_t)(t - 1) * HH + d) * RR + r];
        dst[(size_t)d * 128 + j] = f2bf(v);
    }
}

// ---------------- kernel 4: h = x @ Acat^T (async-LDS staged), scaled by gate*SCALING ----------
__global__ void __launch_bounds__(128) k_h_gemm(const unsigned short* __restrict__ X,
                                                const unsigned short* __restrict__ A,
                                                const float* __restrict__ w,
                                                unsigned short* __restrict__ HW) {
    __shared__ __align__(16) unsigned short btile[2][128 * BROW];
    int tid = threadIdx.x;
    int lane = tid & 31, wid = tid >> 5;
    int row0 = blockIdx.x * 64 + wid * 16;
    FragC c[8];
#pragma unroll
    for (int j = 0; j < 8; ++j)
#pragma unroll
        for (int r = 0; r < 8; ++r) c[j].f[r] = 0.f;

    stage_btile(&btile[0][0], A, 0, 0, tid);
#pragma unroll 1
    for (int k0 = 0; k0 < HH; k0 += 32) {
        int cur = (k0 >> 5) & 1;
        ASYNC_WAIT();
        __syncthreads();
        if (k0 + 32 < HH) stage_btile(&btile[cur ^ 1][0], A, 0, k0 + 32, tid);
        FragAB a;
        load_a(a, X, HH, row0, k0, lane);
        FragAB bs[8];
#pragma unroll
        for (int j = 0; j < 8; ++j) load_b_lds(bs[j], &btile[cur][0], BROW, j * 16, 0, lane);
#pragma unroll
        for (int j = 0; j < 8; ++j) c[j].v = WMMA_BF16(a, bs[j], c[j]);
    }
    int n = lane & 15, hi = lane >> 4;
#pragma unroll
    for (int j = 0; j < 8; ++j)
#pragma unroll
        for (int r = 0; r < 8; ++r) {
            int m = row0 + r + 8 * hi;
            float sc = w[(m >> 10) * TT + j] * 2.0f;  // SCALING = 32/16
            HW[(size_t)m * 128 + j * 16 + n] = f2bf(c[j].f[r] * sc);
        }
}

// ---------------- kernel 5: main GEMM (QKV / output projection) ----------------
// mode 0: bf16 head layout (B,NH,S,HD)  [q,k]
// mode 1: bf16 transposed head layout (B,NH,HD,S)  [v]
// mode 2: fp32 (B,S,H)  [final output]
__global__ void __launch_bounds__(128) k_gemm_main(const unsigned short* __restrict__ X,
                                                   const unsigned short* __restrict__ W,
                                                   const unsigned short* __restrict__ HW,
                                                   const unsigned short* __restrict__ BC,
                                                   unsigned short* __restrict__ outb,
                                                   float* __restrict__ outf, int mode) {
    __shared__ __align__(16) unsigned short btile[2][128 * BROW];
    int tid = threadIdx.x;
    int lane = tid & 31, wid = tid >> 5;
    int row0 = blockIdx.x * 64 + wid * 16;
    int n0 = blockIdx.y * 128;
    FragC c[8];
#pragma unroll
    for (int j = 0; j < 8; ++j)
#pragma unroll
        for (int r = 0; r < 8; ++r) c[j].f[r] = 0.f;

    stage_btile(&btile[0][0], W, n0, 0, tid);
#pragma unroll 1
    for (int k0 = 0; k0 < HH; k0 += 32) {
        int cur = (k0 >> 5) & 1;
        ASYNC_WAIT();
        __syncthreads();  // slab `cur` visible; slab `cur^1` reads (iter-1) retired
        if (k0 + 32 < HH) stage_btile(&btile[cur ^ 1][0], W, n0, k0 + 32, tid);
        FragAB a;
        load_a(a, X, HH, row0, k0, lane);
        FragAB bs[8];
#pragma unroll
        for (int j = 0; j < 8; ++j) load_b_lds(bs[j], &btile[cur][0], BROW, j * 16, 0, lane);
#pragma unroll
        for (int j = 0; j < 8; ++j) c[j].v = WMMA_BF16(a, bs[j], c[j]);
    }
    if (HW) {  // LoRA delta: h (4096x128) @ Bcat^T (2048x128)
#pragma unroll
        for (int k0 = 0; k0 < 128; k0 += 32) {
            FragAB a;
            load_a(a, HW, 128, row0, k0, lane);
            FragAB bs[8];
#pragma unroll
            for (int j = 0; j < 8; ++j) load_b(bs[j], BC, 128, n0 + j * 16, k0, lane);
#pragma unroll
            for (int j = 0; j < 8; ++j) c[j].v = WMMA_BF16(a, bs[j], c[j]);
        }
    }
    int n = lane & 15, hi = lane >> 4;
#pragma unroll
    for (int j = 0; j < 8; ++j)
#pragma unroll
        for (int r = 0; r < 8; ++r) {
            int m = row0 + r + 8 * hi;
            int col = n0 + j * 16 + n;
            float v = c[j].f[r];
            if (mode == 2) {
                outf[(size_t)m * HH + col] = v;
            } else {
                int b = m >> 10, s = m & 1023;
                int head = col >> 7, d = col & 127;
                if (mode == 0)
                    outb[(((size_t)(b * NHD + head)) * SS + s) * HD + d] = f2bf(v);
                else
                    outb[(((size_t)(b * NHD + head)) * HD + d) * SS + s] = f2bf(v);
            }
        }
}

// ---------------- kernel 6: RoPE in place on q,k (B,NH,S,HD) ----------------
__global__ void k_rope(unsigned short* __restrict__ qh, unsigned short* __restrict__ kh,
                       const int* __restrict__ pos_ids) {
    int idx = blockIdx.x * blockDim.x + threadIdx.x;  // B*NH*S*64
    int d = idx & 63;
    int s = (idx >> 6) & (SS - 1);
    int bh = idx >> 16;
    int b = bh >> 4;
    float pos = (float)pos_ids[b * SS + s];
    float inv = __expf(-(float)d * (1.0f / 64.0f) * 9.210340371976184f);  // BASE=1e4
    float ang = pos * inv;
    float cs = __cosf(ang), sn = __sinf(ang);
    size_t base = ((size_t)bh * SS + s) * HD;
    {
        float x1 = bf2f(qh[base + d]), x2 = bf2f(qh[base + d + 64]);
        qh[base + d] = f2bf(x1 * cs - x2 * sn);
        qh[base + d + 64] = f2bf(x2 * cs + x1 * sn);
    }
    {
        float x1 = bf2f(kh[base + d]), x2 = bf2f(kh[base + d + 64]);
        kh[base + d] = f2bf(x1 * cs - x2 * sn);
        kh[base + d + 64] = f2bf(x2 * cs + x1 * sn);
    }
}

// ---------------- kernel 7: flash attention (async-LDS staged K/V) ----------------
__global__ void __launch_bounds__(128) k_flash(const unsigned short* __restrict__ qh,
                                               const unsigned short* __restrict__ kh,
                                               const unsigned short* __restrict__ vt,
                                               unsigned short* __restrict__ attn) {
    __shared__ __align__(16) unsigned short ktile[2][32 * KROW];
    __shared__ __align__(16) unsigned short vtile[2][128 * VROW];
    __shared__ __align__(16) unsigned short pbuf[4 * 16 * 32];
    int tid = threadIdx.x;
    int lane = tid & 31, wid = tid >> 5;
    int bh = blockIdx.y;
    int qb = blockIdx.x * 64 + wid * 16;
    int jmax = blockIdx.x * 64 + 63;  // block-uniform key bound (causal)
    const unsigned short* qbase = qh + (size_t)bh * SS * HD;
    const unsigned short* kbase = kh + (size_t)bh * SS * HD;
    const unsigned short* vbase = vt + (size_t)bh * HD * SS;
    unsigned short* pl = &pbuf[wid * 512];
    FragAB qf[4];
#pragma unroll
    for (int kc = 0; kc < 4; ++kc) load_a(qf[kc], qbase, HD, qb, kc * 32, lane);
    FragC o[8];
#pragma unroll
    for (int j = 0; j < 8; ++j)
#pragma unroll
        for (int r = 0; r < 8; ++r) o[j].f[r] = 0.f;
    float m_i[8], l_i[8];
#pragma unroll
    for (int r = 0; r < 8; ++r) { m_i[r] = -1e30f; l_i[r] = 0.f; }
    int n = lane & 15, hi = lane >> 4;
    const float scale = 0.08838834764831845f;  // 1/sqrt(128)

    stage_ktile(&ktile[0][0], kbase, 0, tid);
    stage_vtile(&vtile[0][0], vbase, 0, tid);
#pragma unroll 1
    for (int j0 = 0; j0 <= jmax; j0 += 32) {
        int cur = (j0 >> 5) & 1;
        ASYNC_WAIT();
        __syncthreads();
        if (j0 + 32 <= jmax) {
            stage_ktile(&ktile[cur ^ 1][0], kbase, j0 + 32, tid);
            stage_vtile(&vtile[cur ^ 1][0], vbase, j0 + 32, tid);
        }
        if (j0 <= qb + 15) {  // wave-uniform: skip fully-masked chunks
            const unsigned short* kt = &ktile[cur][0];
            const unsigned short* vtl = &vtile[cur][0];
            FragAB kb[8];
#pragma unroll
            for (int kc = 0; kc < 4; ++kc) {
                load_b_lds(kb[kc], kt, KROW, 0, kc * 32, lane);
                load_b_lds(kb[4 + kc], kt, KROW, 16, kc * 32, lane);
            }
            FragC s0, s1;
#pragma unroll
            for (int r = 0; r < 8; ++r) { s0.f[r] = 0.f; s1.f[r] = 0.f; }
#pragma unroll
            for (int kc = 0; kc < 4; ++kc) {
                s0.v = WMMA_BF16(qf[kc], kb[kc], s0);
                s1.v = WMMA_BF16(qf[kc], kb[4 + kc], s1);
            }
#pragma unroll
            for (int r = 0; r < 8; ++r) {
                int qrow = qb + r + 8 * hi;
                float v0 = (j0 + n <= qrow) ? s0.f[r] * scale : -1e30f;
                float v1 = (j0 + 16 + n <= qrow) ? s1.f[r] * scale : -1e30f;
                float mx = fmaxf(v0, v1);
#pragma unroll
                for (int off = 1; off < 16; off <<= 1) mx = fmaxf(mx, __shfl_xor(mx, off, 32));
                float mn = fmaxf(m_i[r], mx);
                float alpha = __expf(m_i[r] - mn);
                float e0 = __expf(v0 - mn);
                float e1 = __expf(v1 - mn);
                float sum = e0 + e1;
#pragma unroll
                for (int off = 1; off < 16; off <<= 1) sum += __shfl_xor(sum, off, 32);
                l_i[r] = l_i[r] * alpha + sum;
                m_i[r] = mn;
#pragma unroll
                for (int j = 0; j < 8; ++j) o[j].f[r] *= alpha;
                pl[(r + 8 * hi) * 32 + n] = f2bf(e0);
                pl[(r + 8 * hi) * 32 + 16 + n] = f2bf(e1);
            }
            FragAB pf;
            {
                int m = lane & 15;
                const uint4* pp = (const uint4*)(pl + m * 32 + hi * 8);
                pf.q[0] = pp[0];
                pf.q[1] = pp[2];
            }
            FragAB vf[8];
#pragma unroll
            for (int j = 0; j < 8; ++j) load_b_lds(vf[j], vtl, VROW, j * 16, 0, lane);
#pragma unroll
            for (int j = 0; j < 8; ++j) o[j].v = WMMA_BF16(pf, vf[j], o[j]);
        }
    }
    int b = bh >> 4, head = bh & 15;
#pragma unroll
    for (int j = 0; j < 8; ++j)
#pragma unroll
        for (int r = 0; r < 8; ++r) {
            int qrow = qb + r + 8 * hi;
            float v = o[j].f[r] / l_i[r];
            attn[((size_t)b * SS + qrow) * HH + head * HD + j * 16 + n] = f2bf(v);
        }
}

// ---------------- host ----------------
extern "C" void kernel_launch(void* const* d_in, const int* in_sizes, int n_in,
                              void* d_out, int out_size, void* d_ws, size_t ws_size,
                              hipStream_t stream) {
    (void)in_sizes; (void)n_in; (void)out_size; (void)ws_size;
    const float* x   = (const float*)d_in[0];
    const float* Wq  = (const float*)d_in[1];
    const float* Wk  = (const float*)d_in[2];
    const float* Wv  = (const float*)d_in[3];
    const float* Wo  = (const float*)d_in[4];
    const float* lAq = (const float*)d_in[5];
    const float* lBq = (const float*)d_in[6];
    const float* pAq = (const float*)d_in[7];
    const float* pBq = (const float*)d_in[8];
    const float* lAv = (const float*)d_in[9];
    const float* lBv = (const float*)d_in[10];
    const float* pAv = (const float*)d_in[11];
    const float* pBv = (const float*)d_in[12];
    const float* mq  = (const float*)d_in[13];
    const float* vq  = (const float*)d_in[14];
    const float* mv  = (const float*)d_in[15];
    const float* vv  = (const float*)d_in[16];
    const int*   pos = (const int*)d_in[17];
    float* out = (float*)d_out;

    char* ws = (char*)d_ws;
    size_t off = 0;
    auto alloc = [&](size_t bytes) -> char* {
        char* p = ws + off;
        off = (off + bytes + 255) & ~(size_t)255;
        return p;
    };
    const size_t MTOT = (size_t)NB * SS;  // 4096
    float* wgq = (float*)alloc(NB * TT * 4);
    float* wgv = (float*)alloc(NB * TT * 4);
    unsigned short* xb   = (unsigned short*)alloc(MTOT * HH * 2);
    unsigned short* Wqb  = (unsigned short*)alloc((size_t)HH * HH * 2);
    unsigned short* Wkb  = (unsigned short*)alloc((size_t)HH * HH * 2);
    unsigned short* Wvb  = (unsigned short*)alloc((size_t)HH * HH * 2);
    unsigned short* Wob  = (unsigned short*)alloc((size_t)HH * HH * 2);
    unsigned short* Aq   = (unsigned short*)alloc((size_t)128 * HH * 2);
    unsigned short* Av   = (unsigned short*)alloc((size_t)128 * HH * 2);
    unsigned short* Bq   = (unsigned short*)alloc((size_t)HH * 128 * 2);
    unsigned short* Bv   = (unsigned short*)alloc((size_t)HH * 128 * 2);
    unsigned short* hWq  = (unsigned short*)alloc(MTOT * 128 * 2);
    unsigned short* hWv  = (unsigned short*)alloc(MTOT * 128 * 2);
    unsigned short* qhb  = (unsigned short*)alloc(MTOT * HH * 2);
    unsigned short* khb  = (unsigned short*)alloc(MTOT * HH * 2);
    unsigned short* vtb  = (unsigned short*)alloc(MTOT * HH * 2);
    unsigned short* attn = (unsigned short*)alloc(MTOT * HH * 2);

    k_gate<<<dim3(NB), 256, 0, stream>>>(x, mq, vq, mv, vv, wgq, wgv);
    k_pack<<<4096, 256, 0, stream>>>(x, xb, (int)(MTOT * HH));
    k_pack<<<2048, 256, 0, stream>>>(Wq, Wqb, HH * HH);
    k_pack<<<2048, 256, 0, stream>>>(Wk, Wkb, HH * HH);
    k_pack<<<2048, 256, 0, stream>>>(Wv, Wvb, HH * HH);
    k_pack<<<2048, 256, 0, stream>>>(Wo, Wob, HH * HH);
    k_build_cats<<<dim3(1024, 4), 256, 0, stream>>>(lAq, pAq, lBq, pBq, lAv, pAv, lBv, pBv,
                                                    Aq, Av, Bq, Bv);
    k_h_gemm<<<64, 128, 0, stream>>>(xb, Aq, wgq, hWq);
    k_h_gemm<<<64, 128, 0, stream>>>(xb, Av, wgv, hWv);
    k_gemm_main<<<dim3(64, 16), 128, 0, stream>>>(xb, Wqb, hWq, Bq, qhb, nullptr, 0);
    k_gemm_main<<<dim3(64, 16), 128, 0, stream>>>(xb, Wkb, nullptr, nullptr, khb, nullptr, 0);
    k_gemm_main<<<dim3(64, 16), 128, 0, stream>>>(xb, Wvb, hWv, Bv, vtb, nullptr, 1);
    k_rope<<<16384, 256, 0, stream>>>(qhb, khb, pos);
    k_flash<<<dim3(SS / 64, NB * NHD), 128, 0, stream>>>(qhb, khb, vtb, attn);
    k_gemm_main<<<dim3(64, 16), 128, 0, stream>>>(attn, Wob, nullptr, nullptr, nullptr, out, 2);
}
